// CriticUAVob_81509889344036
// MI455X (gfx1250) — matrix operand order repaired and checked by reading, not compiled, not run on hardware
//
#include <hip/hip_runtime.h>
#include <hip/hip_bf16.h>

typedef __attribute__((ext_vector_type(16))) _Float16 v16h;
typedef __attribute__((ext_vector_type(8)))  _Float16 v8h;
typedef __attribute__((ext_vector_type(8)))  float    v8f;

#define NSEQ 128
#define DQK  16
#define DVV  32
#define DHID 128
#define KPADW 32   // K stored padded to 32 cols; cols 16..31 are real zeros
#define VTPAD 136  // V^T row stride in halfs: 272 B (16B multiple, bank-staggered)
#define APAD  72   // staging row stride in halfs: 144 B (16B multiple)

__global__ __launch_bounds__(256) void critic_attn_kernel(
    const float* __restrict__ s_obs,
    const float* __restrict__ Wq0, const float* __restrict__ bq0,
    const float* __restrict__ Wk0, const float* __restrict__ bk0,
    const float* __restrict__ Wv0, const float* __restrict__ bv0,
    const float* __restrict__ Wq1, const float* __restrict__ bq1,
    const float* __restrict__ Wk1, const float* __restrict__ bk1,
    const float* __restrict__ Wv1, const float* __restrict__ bv1,
    const float* __restrict__ W1,  const float* __restrict__ b1,
    const float* __restrict__ W2,  const float* __restrict__ b2,
    const float* __restrict__ W3,  const float* __restrict__ b3,
    float* __restrict__ out)
{
    __shared__ float                sS[NSEQ][3];              // 1.5 KB
    __shared__ __align__(16) _Float16 sQ[2][NSEQ][DQK];       // 8 KB (32B rows)
    __shared__ __align__(32) _Float16 sKp[2][NSEQ][KPADW];    // 16 KB, zero-padded K
    __shared__ __align__(32) _Float16 sVt[2][DVV][VTPAD];     // 17 KB, V transposed
    __shared__ __align__(16) _Float16 sAttn[8][16][APAD];     // 18 KB staging
    __shared__ float                sH[2 * DVV];
    __shared__ float                sHid[DHID];
    __shared__ float                sHid2[DHID];

    const int tid  = threadIdx.x;
    const int b    = blockIdx.x;
    const int wave = tid >> 5;
    const int lane = tid & 31;
    const int l16  = lane & 15;
    const bool hi  = (lane >= 16);

    const float* Wq[2] = {Wq0, Wq1};  const float* bq[2] = {bq0, bq1};
    const float* Wk[2] = {Wk0, Wk1};  const float* bk[2] = {bk0, bk1};
    const float* Wv[2] = {Wv0, Wv1};  const float* bv[2] = {bv0, bv1};

    // ---------------- stage 0: load obs, zero K pad, project Q/K/V ----------
    const float* sb = s_obs + (size_t)b * NSEQ * 3;
    for (int i = tid; i < NSEQ * 3; i += 256) sS[i / 3][i % 3] = sb[i];
    // zero the K pad region (cols 16..31 of every row): 2*128 rows * 2 int4
    {
        int4* z4 = (int4*)&sKp[0][0][0];
        for (int i = tid; i < 2 * NSEQ * 2; i += 256) {
            const int row = i >> 1, h = i & 1;
            z4[row * 4 + 2 + h] = make_int4(0, 0, 0, 0);
        }
    }
    if (tid < 2 * DVV) sH[tid] = 0.0f;
    __syncthreads();

    #pragma unroll
    for (int p = 0; p < 2; ++p) {
        for (int e = tid; e < NSEQ * 64; e += 256) {
            const int row = e >> 6;
            const int c   = e & 63;
            const float s0 = sS[row][0], s1 = sS[row][1], s2 = sS[row][2];
            if (c < DQK) {
                float v = bq[p][c] + s0 * Wq[p][c] + s1 * Wq[p][DQK + c]
                                   + s2 * Wq[p][2 * DQK + c];
                sQ[p][row][c] = (_Float16)v;
            } else if (c < 2 * DQK) {
                const int cc = c - DQK;
                float v = bk[p][cc] + s0 * Wk[p][cc] + s1 * Wk[p][DQK + cc]
                                    + s2 * Wk[p][2 * DQK + cc];
                sKp[p][row][cc] = (_Float16)v;          // cols 16..31 stay zero
            } else {
                const int cc = c - 2 * DQK;
                float v = bv[p][cc] + s0 * Wv[p][cc] + s1 * Wv[p][DVV + cc]
                                    + s2 * Wv[p][2 * DVV + cc];
                sVt[p][cc][row] = (_Float16)v;          // transposed store
            }
        }
    }
    __syncthreads();

    // --------- stage 1: per wave, one 16-row block per pool ------------------
    const float c_scale = 0.25f * 1.44269504088896f;   // 1/sqrt(16) * log2(e)
    const v8h  zro8 = {(_Float16)0, (_Float16)0, (_Float16)0, (_Float16)0,
                       (_Float16)0, (_Float16)0, (_Float16)0, (_Float16)0};

    #pragma unroll
    for (int p = 0; p < 2; ++p) {
        const int qb = wave;                     // 16-row block id, 0..7

        // ---- A fragment = Q rows (16x32 f16, K zero-padded 16->32) ----------
        // lanes 0-15: elems 0..7 = K0..7, 8..15 = K16..23 (zeros);
        // lanes 16-31: elems 0..7 = K8..15, 8..15 = K24..31 (zeros)
        v16h afrag;
        {
            const int row = qb * 16 + l16;
            const v8h qlo = *(const v8h*)&sQ[p][row][hi ? 8 : 0];  // one b128
            afrag = __builtin_shufflevector(qlo, zro8,
                     0, 1, 2, 3, 4, 5, 6, 7, 8, 9, 10, 11, 12, 13, 14, 15);
        }

        // ---- scores: 8 tiles of 16x16 across the 128 keys -------------------
        v8f acc[8];
        #pragma unroll
        for (int mt = 0; mt < 8; ++mt) {
            // B[k][m] = K[m][k]; lanes 0-15 -> elems 0..7 = K0..7,
            // lanes 16-31 -> elems 0..7 = K8..15; elems 8..15 = K16..31 (zeros)
            const int m = mt * 16 + l16;
            const v16h bfrag = __builtin_shufflevector(*(const v8h*)&sKp[p][m][hi ? 8 : 0], zro8,
                     0, 1, 2, 3, 4, 5, 6, 7, 8, 9, 10, 11, 12, 13, 14, 15); // one b128
            v8f z = {0.f, 0.f, 0.f, 0.f, 0.f, 0.f, 0.f, 0.f};
            acc[mt] = __builtin_amdgcn_wmma_f32_16x16x32_f16(
                false, afrag, false, bfrag, (short)0, z, false, false);
        }

        // ---- in-register softmax over 128 keys ------------------------------
        #pragma unroll
        for (int r = 0; r < 8; ++r) {
            float mx = acc[0][r];
            #pragma unroll
            for (int t = 1; t < 8; ++t) mx = fmaxf(mx, acc[t][r]);
            #pragma unroll
            for (int off = 1; off < 16; off <<= 1)
                mx = fmaxf(mx, __shfl_xor(mx, off, 16));
            float sum = 0.0f;
            #pragma unroll
            for (int t = 0; t < 8; ++t) {
                const float pz = __builtin_exp2f((acc[t][r] - mx) * c_scale);
                acc[t][r] = pz;
                sum += pz;
            }
            #pragma unroll
            for (int off = 1; off < 16; off <<= 1)
                sum += __shfl_xor(sum, off, 16);
            const float inv = 1.0f / sum;
            #pragma unroll
            for (int t = 0; t < 8; ++t) acc[t][r] *= inv;
        }

        // ---- attn @ V : D-layout -> LDS staging -> A-layout, 2 halves -------
        v8f oacc[2];
        #pragma unroll
        for (int i = 0; i < 8; ++i) { oacc[0][i] = 0.f; oacc[1][i] = 0.f; }
        const int myrow = hi ? 8 : 0;

        #pragma unroll
        for (int half = 0; half < 2; ++half) {
            #pragma unroll
            for (int t = 0; t < 4; ++t) {
                const int col = t * 16 + l16;
                #pragma unroll
                for (int r = 0; r < 8; ++r)
                    sAttn[wave][myrow + r][col] = (_Float16)acc[half * 4 + t][r];
            }
            // same-wave DS ops are in order; just stop the compiler reordering
            asm volatile("" ::: "memory");

            #pragma unroll
            for (int kc2 = 0; kc2 < 2; ++kc2) {
                const int kc = half * 2 + kc2;      // K chunk over the 128 keys
                const int basec = kc2 * 32 + (hi ? 16 : 0);
                const v8h alo = *(const v8h*)&sAttn[wave][l16][basec];      // b128
                const v8h ahi = *(const v8h*)&sAttn[wave][l16][basec + 8];  // b128
                const v16h aa = __builtin_shufflevector(alo, ahi,
                     0, 1, 2, 3, 4, 5, 6, 7, 8, 9, 10, 11, 12, 13, 14, 15);
                #pragma unroll
                for (int nt = 0; nt < 2; ++nt) {
                    // B[k][n] = V[kc*32+k][n] = Vt[n][kc*32+k] : contiguous
                    const int n    = nt * 16 + l16;
                    const int koff = kc * 32 + (hi ? 16 : 0);
                    const v16h bb = *(const v16h*)&sVt[p][n][koff];  // 2x b128
                    oacc[nt] = __builtin_amdgcn_wmma_f32_16x16x32_f16(
                        false, aa, false, bb, (short)0, oacc[nt], false, false);
                }
            }
        }

        // ---- mean pool: column sums of this wave's 16 rows ------------------
        #pragma unroll
        for (int nt = 0; nt < 2; ++nt) {
            float cs = 0.0f;
            #pragma unroll
            for (int r = 0; r < 8; ++r) cs += oacc[nt][r];
            cs += __shfl_xor(cs, 16, 32);           // fold M=0..7 with M=8..15
            if (!hi) atomicAdd(&sH[p * DVV + nt * 16 + l16], cs);  // ds_add_f32
        }
    }
    __syncthreads();

    // ---------------- stage 2: tiny MLP head --------------------------------
    const float invN = 1.0f / (float)NSEQ;
    if (tid < DHID) {
        float a1 = b1[tid];
        #pragma unroll
        for (int i = 0; i < 2 * DVV; ++i)
            a1 += (sH[i] * invN) * W1[i * DHID + tid];
        sHid[tid] = tanhf(a1);
    }
    __syncthreads();
    if (tid < DHID) {
        float a2 = b2[tid];
        for (int i = 0; i < DHID; ++i)
            a2 += sHid[i] * W2[i * DHID + tid];
        sHid2[tid] = tanhf(a2);
    }
    __syncthreads();
    if (tid < 32) {
        float accf = 0.0f;
        for (int j = tid; j < DHID; j += 32) accf += sHid2[j] * W3[j];
        #pragma unroll
        for (int off = 16; off >= 1; off >>= 1)
            accf += __shfl_xor(accf, off, 32);
        if (tid == 0) out[b] = accf + b3[0];
    }
}

extern "C" void kernel_launch(void* const* d_in, const int* in_sizes, int n_in,
                              void* d_out, int out_size, void* d_ws, size_t ws_size,
                              hipStream_t stream) {
    (void)in_sizes; (void)n_in; (void)d_ws; (void)ws_size; (void)out_size;
    critic_attn_kernel<<<4096, 256, 0, stream>>>(
        (const float*)d_in[0],
        (const float*)d_in[1],  (const float*)d_in[2],
        (const float*)d_in[3],  (const float*)d_in[4],
        (const float*)d_in[5],  (const float*)d_in[6],
        (const float*)d_in[7],  (const float*)d_in[8],
        (const float*)d_in[9],  (const float*)d_in[10],
        (const float*)d_in[11], (const float*)d_in[12],
        (const float*)d_in[13], (const float*)d_in[14],
        (const float*)d_in[15], (const float*)d_in[16],
        (const float*)d_in[17], (const float*)d_in[18],
        (float*)d_out);
}
